// GQARotaryDecoder_55740085567783
// MI455X (gfx1250) — matrix operand order repaired and checked by reading, not compiled, not run on hardware
//
#include <hip/hip_runtime.h>
#include <hip/hip_bf16.h>

// ---------------- problem constants ----------------
#define HIDDEN    4096
#define Q_HEADS   32
#define KV_HEADS  8
#define HEAD_DIM  128
#define MAX_LEN   8192
#define PAST_LEN  4096
#define SEQ_LEN   16
#define QKV_DIM   ((Q_HEADS + 2 * KV_HEADS) * HEAD_DIM)   // 6144
#define KEY_LEN   (PAST_LEN + SEQ_LEN)                    // 4112
#define NCHUNK    4                                       // split-K factor

typedef __attribute__((ext_vector_type(16))) __bf16 v16bf;
typedef __attribute__((ext_vector_type(8)))  __bf16 v8bf;
typedef __attribute__((ext_vector_type(8)))  float  v8f;
typedef __attribute__((ext_vector_type(4)))  float  v4f;

static __device__ __forceinline__ v8f wmma_bf16(const v16bf& a, const v16bf& b, const v8f& c) {
  return __builtin_amdgcn_wmma_f32_16x16x32_bf16(false, a, false, b, (short)0, c, false, false);
}

// ---------------- f32 -> bf16 convert ----------------
__global__ void cvt_bf16_kernel(const float* __restrict__ in, __bf16* __restrict__ out, int n) {
  int i = blockIdx.x * blockDim.x + threadIdx.x;
  if (i < n) out[i] = (__bf16)in[i];
}

// ---------------- WMMA GEMM: out[16][N] = A[16][K] * W[N][K]^T, split-K ----------------
// A: bf16 row-major [16][K]; W: f32 row-major [N][K]; partial: f32 [NCHUNK][16][N]
__global__ void gemm16_wmma_kernel(const __bf16* __restrict__ A,
                                   const float*  __restrict__ W,
                                   float*        __restrict__ partial,
                                   int K, int N, int kChunk) {
  const int lane  = threadIdx.x & 31;
  const int wave  = threadIdx.x >> 5;
  const int nTile = blockIdx.x * (blockDim.x >> 5) + wave;
  const int n0    = nTile * 16;
  if (n0 >= N) return;
  const int chunk   = blockIdx.y;
  const int k_begin = chunk * kChunk;
  const int k_end   = k_begin + kChunk;

  const int hi   = lane >> 4;      // half-wave select
  const int nl   = lane & 15;
  const int koff = hi * 8;         // A-fragment K sub-offset per ISA layout

  const __bf16* arow = A + (size_t)nl * K;             // A row m = lane&15
  const float*  wrow = W + (size_t)(n0 + nl) * K;      // B col n = lane&15 -> W row

  v8f c = {};
  for (int k0 = k_begin; k0 < k_end; k0 += 32) {
    // ---- A fragment: two contiguous 8-element bf16 runs (this code's contiguous order k = 16*hi + s, same as B)
    v16bf a;
    {
      v8bf a0 = *(const v8bf*)(arow + k0 + hi * 16);
      v8bf a1 = *(const v8bf*)(arow + k0 + hi * 16 + 8);
#pragma unroll
      for (int j = 0; j < 8; ++j) { a[j] = a0[j]; a[8 + j] = a1[j]; }
    }
    // ---- B fragment: 16 consecutive f32 of one weight row (64B, coalesces to 128B/row pair)
    v16bf b;
    {
      const v4f* q = (const v4f*)(wrow + k0 + hi * 16);
      v4f w0 = q[0], w1 = q[1], w2 = q[2], w3 = q[3];
#pragma unroll
      for (int j = 0; j < 4; ++j) {
        b[j]      = (__bf16)w0[j];
        b[4 + j]  = (__bf16)w1[j];
        b[8 + j]  = (__bf16)w2[j];
        b[12 + j] = (__bf16)w3[j];
      }
    }
    __builtin_prefetch(wrow + k0 + 256, 0, 1);   // global_prefetch_b8: stream-ahead 1KB
    c = wmma_bf16(a, b, c);
  }

  // C layout: element r -> row m = r + hi*8, col = n0 + (lane&15)
  float* orow = partial + (size_t)chunk * 16 * N;
#pragma unroll
  for (int r = 0; r < 8; ++r)
    orow[(size_t)(r + hi * 8) * N + n0 + nl] = c[r];
}

__global__ void reduce_chunks_kernel(const float* __restrict__ partial,
                                     float* __restrict__ out, int n) {
  int i = blockIdx.x * blockDim.x + threadIdx.x;
  if (i >= n) return;
  float s = 0.f;
#pragma unroll
  for (int c = 0; c < NCHUNK; ++c) s += partial[(size_t)c * n + i];
  out[i] = s;
}

// ---------------- RoPE (reference: positions 0..15, inv_freq = d/half, LINEAR) ----------------
__global__ void rope_kernel(const float* __restrict__ qkv,      // [16][6144]
                            float* __restrict__ q_rot,          // [32][16][128]
                            float* __restrict__ k_rot) {        // [8][16][128]
  int idx = blockIdx.x * blockDim.x + threadIdx.x;
  if (idx >= (Q_HEADS + KV_HEADS) * SEQ_LEN * 64) return;
  int d  = idx & 63;
  int s  = (idx >> 6) & 15;
  int hh = idx >> 10;                            // 0..39: 0-31 q heads, 32-39 k heads
  float ang = (float)s * ((float)d * (1.0f / 64.0f));
  float sn = __sinf(ang), cs = __cosf(ang);
  int o = (hh < Q_HEADS) ? hh * HEAD_DIM
                         : Q_HEADS * HEAD_DIM + (hh - Q_HEADS) * HEAD_DIM;
  float xe = qkv[s * QKV_DIM + o + d];
  float xo = qkv[s * QKV_DIM + o + d + 64];
  float lo = xe * cs - xo * sn;
  float hi = xe * sn + xo * cs;
  float* dst = (hh < Q_HEADS)
      ? q_rot + ((size_t)hh * SEQ_LEN + s) * HEAD_DIM
      : k_rot + ((size_t)(hh - Q_HEADS) * SEQ_LEN + s) * HEAD_DIM;
  dst[d] = lo;
  dst[d + 64] = hi;
}

__global__ void vcopy_kernel(const float* __restrict__ qkv, float* __restrict__ v_new) {
  int idx = blockIdx.x * blockDim.x + threadIdx.x;         // 8*16*128
  if (idx >= KV_HEADS * SEQ_LEN * HEAD_DIM) return;
  int d  = idx & 127;
  int s  = (idx >> 7) & 15;
  int kv = idx >> 11;
  v_new[idx] = qkv[s * QKV_DIM + (Q_HEADS + KV_HEADS) * HEAD_DIM + kv * HEAD_DIM + d];
}

// ---------------- flash-style GQA attention, 1 block (4 waves) per q head ----------------
__global__ void __launch_bounds__(128)
attn_kernel(const float* __restrict__ q_rot,    // [32][16][128]
            const float* __restrict__ cache_k,  // [8][8192][128]
            const float* __restrict__ cache_v,  // [8][8192][128]
            const float* __restrict__ k_rot,    // [8][16][128]
            const float* __restrict__ v_new,    // [8][16][128]
            float* __restrict__ attn_out) {     // [16][4096]  (o = h*128+d)
  const int h    = blockIdx.x;
  const int kv   = h >> 2;                       // REPEAT_KV = 4
  const int lane = threadIdx.x & 31;
  const int wave = threadIdx.x >> 5;             // owns d-cols [wave*32, wave*32+32)
  const int hi   = lane >> 4;
  const int nl   = lane & 15;
  const int koff = hi * 8;
  const float SCALE = 0.08838834764831845f;      // 128^-0.5

  __shared__ __attribute__((aligned(16))) __bf16 lds_p[4][16 * 32];
  __bf16* Lp = lds_p[wave];

  // ---- q fragments (A-matrix, 4 K-steps of 32 over HEAD_DIM), kept in registers
  v16bf aq[4];
  {
    const float* qrow = q_rot + ((size_t)h * SEQ_LEN + nl) * HEAD_DIM;
#pragma unroll
    for (int kk = 0; kk < 4; ++kk) {
      const v4f* p0 = (const v4f*)(qrow + kk * 32 + hi * 16);
      const v4f* p1 = (const v4f*)(qrow + kk * 32 + hi * 16 + 8);
      v4f x0 = p0[0], x1 = p0[1], y0 = p1[0], y1 = p1[1];
#pragma unroll
      for (int j = 0; j < 4; ++j) {
        aq[kk][j]      = (__bf16)x0[j];
        aq[kk][4 + j]  = (__bf16)x1[j];
        aq[kk][8 + j]  = (__bf16)y0[j];
        aq[kk][12 + j] = (__bf16)y1[j];
      }
    }
  }

  float rmax[8], rsum[8];
#pragma unroll
  for (int r = 0; r < 8; ++r) { rmax[r] = -1e30f; rsum[r] = 0.f; }
  v8f acc0 = {}, acc1 = {};
  const int d0 = wave * 32;

  // ================= main loop: 128 blocks of 32 past keys =================
  for (int t0 = 0; t0 < PAST_LEN; t0 += 32) {
    // ---- S = q K^T  (two 16-key tiles)
    v8f s0 = {}, s1 = {};
#pragma unroll
    for (int kk = 0; kk < 4; ++kk) {
      const float* kr0 = cache_k + ((size_t)kv * MAX_LEN + t0 + nl) * HEAD_DIM + kk * 32 + hi * 16;
      const float* kr1 = kr0 + 16 * HEAD_DIM;
      v16bf b0, b1;
      {
        const v4f* q0 = (const v4f*)kr0; const v4f* q1 = (const v4f*)kr1;
        v4f a_ = q0[0], b_ = q0[1], c_ = q0[2], d_ = q0[3];
        v4f e_ = q1[0], f_ = q1[1], g_ = q1[2], h_ = q1[3];
#pragma unroll
        for (int j = 0; j < 4; ++j) {
          b0[j] = (__bf16)a_[j]; b0[4 + j] = (__bf16)b_[j];
          b0[8 + j] = (__bf16)c_[j]; b0[12 + j] = (__bf16)d_[j];
          b1[j] = (__bf16)e_[j]; b1[4 + j] = (__bf16)f_[j];
          b1[8 + j] = (__bf16)g_[j]; b1[12 + j] = (__bf16)h_[j];
        }
      }
      s0 = wmma_bf16(aq[kk], b0, s0);
      s1 = wmma_bf16(aq[kk], b1, s1);
    }

    // ---- online softmax (row reductions across the 16-lane half via shfl_xor)
#pragma unroll
    for (int r = 0; r < 8; ++r) {
      float a = s0[r] * SCALE, b = s1[r] * SCALE;
      float mx = fmaxf(a, b);
#pragma unroll
      for (int m_ = 1; m_ < 16; m_ <<= 1) mx = fmaxf(mx, __shfl_xor(mx, m_, 32));
      float nm = fmaxf(rmax[r], mx);
      float alpha = __expf(rmax[r] - nm);
      float p0 = __expf(a - nm), p1 = __expf(b - nm);
      float ps = p0 + p1;
#pragma unroll
      for (int m_ = 1; m_ < 16; m_ <<= 1) ps += __shfl_xor(ps, m_, 32);
      rsum[r] = rsum[r] * alpha + ps;
      rmax[r] = nm;
      acc0[r] *= alpha; acc1[r] *= alpha;
      int m = r + hi * 8;
      Lp[m * 32 + nl]      = (__bf16)p0;   // C-layout -> LDS
      Lp[m * 32 + 16 + nl] = (__bf16)p1;
    }

    // ---- reload P in A-layout (DS ops in-order within the wave)
    v16bf ap;
    {
      v8bf r0 = *(const v8bf*)(Lp + nl * 32 + hi * 16);
      v8bf r1 = *(const v8bf*)(Lp + nl * 32 + hi * 16 + 8);
#pragma unroll
      for (int j = 0; j < 8; ++j) { ap[j] = r0[j]; ap[8 + j] = r1[j]; }
    }

    // ---- O += P V   (two 16-wide head-dim tiles per wave)
    {
      const float* vb = cache_v + ((size_t)kv * MAX_LEN + t0 + hi * 16) * HEAD_DIM + d0 + nl;
      v16bf bv0, bv1;
#pragma unroll
      for (int j = 0; j < 16; ++j) {
        bv0[j] = (__bf16)vb[(size_t)j * HEAD_DIM];
        bv1[j] = (__bf16)vb[(size_t)j * HEAD_DIM + 16];
      }
      acc0 = wmma_bf16(ap, bv0, acc0);
      acc1 = wmma_bf16(ap, bv1, acc1);
    }
  }

  // ================= tail: 16 new keys (RoPE'd), causal mask =================
  {
    v8f s0 = {};
#pragma unroll
    for (int kk = 0; kk < 4; ++kk) {
      const float* kr = k_rot + ((size_t)kv * SEQ_LEN + nl) * HEAD_DIM + kk * 32 + hi * 16;
      v16bf b0;
      const v4f* q0 = (const v4f*)kr;
      v4f a_ = q0[0], b_ = q0[1], c_ = q0[2], d_ = q0[3];
#pragma unroll
      for (int j = 0; j < 4; ++j) {
        b0[j] = (__bf16)a_[j]; b0[4 + j] = (__bf16)b_[j];
        b0[8 + j] = (__bf16)c_[j]; b0[12 + j] = (__bf16)d_[j];
      }
      s0 = wmma_bf16(aq[kk], b0, s0);
    }
#pragma unroll
    for (int r = 0; r < 8; ++r) {
      int m = r + hi * 8;
      float a = s0[r] * SCALE + ((nl <= m) ? 0.f : -10000.f);  // causal: key 4096+n vs query m+4096
      float mx = a;
#pragma unroll
      for (int m_ = 1; m_ < 16; m_ <<= 1) mx = fmaxf(mx, __shfl_xor(mx, m_, 32));
      float nm = fmaxf(rmax[r], mx);
      float alpha = __expf(rmax[r] - nm);
      float p0 = __expf(a - nm);
      float ps = p0;
#pragma unroll
      for (int m_ = 1; m_ < 16; m_ <<= 1) ps += __shfl_xor(ps, m_, 32);
      rsum[r] = rsum[r] * alpha + ps;
      rmax[r] = nm;
      acc0[r] *= alpha; acc1[r] *= alpha;
      Lp[m * 32 + nl]      = (__bf16)p0;
      Lp[m * 32 + 16 + nl] = (__bf16)0.f;   // padded keys 16..31 contribute nothing
    }
    v16bf ap;
    {
      v8bf r0 = *(const v8bf*)(Lp + nl * 32 + hi * 16);
      v8bf r1 = *(const v8bf*)(Lp + nl * 32 + hi * 16 + 8);
#pragma unroll
      for (int j = 0; j < 8; ++j) { ap[j] = r0[j]; ap[8 + j] = r1[j]; }
    }
    {
      const float* vb = v_new + ((size_t)kv * SEQ_LEN + hi * 16) * HEAD_DIM + d0 + nl;
      v16bf bv0, bv1;
#pragma unroll
      for (int j = 0; j < 16; ++j) {
        float v0 = (hi == 0) ? vb[(size_t)j * HEAD_DIM] : 0.f;       // keys 16..31 are padding
        float v1 = (hi == 0) ? vb[(size_t)j * HEAD_DIM + 16] : 0.f;
        bv0[j] = (__bf16)v0; bv1[j] = (__bf16)v1;
      }
      acc0 = wmma_bf16(ap, bv0, acc0);
      acc1 = wmma_bf16(ap, bv1, acc1);
    }
  }

  // ---- normalize + store: out[s][h*128 + d]
#pragma unroll
  for (int r = 0; r < 8; ++r) {
    int s = r + hi * 8;
    float inv = 1.0f / rsum[r];
    attn_out[(size_t)s * (Q_HEADS * HEAD_DIM) + h * HEAD_DIM + d0 + nl]      = acc0[r] * inv;
    attn_out[(size_t)s * (Q_HEADS * HEAD_DIM) + h * HEAD_DIM + d0 + 16 + nl] = acc1[r] * inv;
  }
}

// ---------------- host-side orchestration ----------------
extern "C" void kernel_launch(void* const* d_in, const int* in_sizes, int n_in,
                              void* d_out, int out_size, void* d_ws, size_t ws_size,
                              hipStream_t stream) {
  const float* x       = (const float*)d_in[0];   // [1][16][4096]
  const float* w_qkv   = (const float*)d_in[1];   // [6144][4096]
  const float* w_out   = (const float*)d_in[2];   // [4096][4096]
  const float* cache_k = (const float*)d_in[3];   // [1][8][8192][128]
  const float* cache_v = (const float*)d_in[4];
  float* out = (float*)d_out;                     // [1][16][4096]

  // workspace layout (bytes)
  char* ws = (char*)d_ws;
  size_t off = 0;
  auto alloc = [&](size_t bytes) { char* p = ws + off; off += (bytes + 255) & ~size_t(255); return p; };
  __bf16* x_bf16    = (__bf16*)alloc(SEQ_LEN * HIDDEN * sizeof(__bf16));
  float*  qkv       = (float*) alloc(SEQ_LEN * QKV_DIM * sizeof(float));
  float*  q_rot     = (float*) alloc((size_t)Q_HEADS * SEQ_LEN * HEAD_DIM * sizeof(float));
  float*  k_rot     = (float*) alloc((size_t)KV_HEADS * SEQ_LEN * HEAD_DIM * sizeof(float));
  float*  v_new     = (float*) alloc((size_t)KV_HEADS * SEQ_LEN * HEAD_DIM * sizeof(float));
  float*  attn_o    = (float*) alloc((size_t)SEQ_LEN * HIDDEN * sizeof(float));
  __bf16* attn_bf16 = (__bf16*)alloc(SEQ_LEN * HIDDEN * sizeof(__bf16));
  float*  partial   = (float*) alloc((size_t)NCHUNK * SEQ_LEN * QKV_DIM * sizeof(float));

  // 1) x -> bf16
  cvt_bf16_kernel<<<(SEQ_LEN * HIDDEN + 255) / 256, 256, 0, stream>>>(x, x_bf16, SEQ_LEN * HIDDEN);
  // 2) qkv = x * w_qkv^T   (split-K=4 -> 192 blocks)
  gemm16_wmma_kernel<<<dim3(QKV_DIM / 128, NCHUNK), 256, 0, stream>>>(
      x_bf16, w_qkv, partial, HIDDEN, QKV_DIM, HIDDEN / NCHUNK);
  reduce_chunks_kernel<<<(SEQ_LEN * QKV_DIM + 255) / 256, 256, 0, stream>>>(
      partial, qkv, SEQ_LEN * QKV_DIM);
  // 3) RoPE q/k + copy v
  rope_kernel<<<((Q_HEADS + KV_HEADS) * SEQ_LEN * 64 + 255) / 256, 256, 0, stream>>>(
      qkv, q_rot, k_rot);
  vcopy_kernel<<<(KV_HEADS * SEQ_LEN * HEAD_DIM + 255) / 256, 256, 0, stream>>>(qkv, v_new);
  // 4) attention (32 q heads, 4 waves each)
  attn_kernel<<<Q_HEADS, 128, 0, stream>>>(q_rot, cache_k, cache_v, k_rot, v_new, attn_o);
  // 5) out = attn * w_out^T
  cvt_bf16_kernel<<<(SEQ_LEN * HIDDEN + 255) / 256, 256, 0, stream>>>(attn_o, attn_bf16, SEQ_LEN * HIDDEN);
  gemm16_wmma_kernel<<<dim3(HIDDEN / 128, NCHUNK), 256, 0, stream>>>(
      attn_bf16, w_out, partial, HIDDEN, HIDDEN, HIDDEN / NCHUNK);
  reduce_chunks_kernel<<<(SEQ_LEN * HIDDEN + 255) / 256, 256, 0, stream>>>(
      partial, out, SEQ_LEN * HIDDEN);
}
